// CustomGraphNet_42743514530392
// MI455X (gfx1250) — compile-verified, matrix operand order checked by reading, not run on hardware
//
#include <hip/hip_runtime.h>
#include <math.h>

#define NN   50000
#define NE   1600000
#define NG   50
#define TSEQ 128

typedef __attribute__((ext_vector_type(16))) _Float16 v16h;
typedef __attribute__((ext_vector_type(8)))  float    v8f;

__device__ __forceinline__ float lrelu_(float v) { return v > 0.f ? v : 0.01f * v; }
__device__ __forceinline__ unsigned f2ord(float f) {
  unsigned u = __float_as_uint(f);
  return (u & 0x80000000u) ? ~u : (u | 0x80000000u);
}
__device__ __forceinline__ float ord2f(unsigned o) {
  unsigned b = (o & 0x80000000u) ? (o & 0x7fffffffu) : ~o;
  return __uint_as_float(b);
}

// ---------------------------------------------------------------------------
// Generic zeroing (d_ws is poisoned; deterministic init each call)
// ---------------------------------------------------------------------------
__global__ void k_zero(float* __restrict__ p, long n) {
  long i = (long)blockIdx.x * blockDim.x + threadIdx.x;
  long stride = (long)gridDim.x * blockDim.x;
  for (; i < n; i += stride) p[i] = 0.f;
}

// ---------------------------------------------------------------------------
// WMMA tiled GEMM:  C[M,N] = act( A[M,K] @ W[N,K]^T + bias + resid )
// K is compile-time (16 or 64) so operand loads are unpredicated contiguous
// 32B runs (mergeable into global_load_b128) and K-padding is constant-folded.
// Row/col clamping uses pointer clamp only: in D = A x B, row r of D depends
// only on row r of A and col n only on col n of B, and out-of-range rows/cols
// are never stored, so no data zeroing is needed (only K must be exact).
// A-frag (ISA 7.12.2, 16-bit A 16x32): lanes 0-15 row=lane, K {0..7,16..23};
// lanes 16-31 row=lane-16, K {8..15,24..31}. B symmetric (K-striped).
// C/D: lane<16 -> N=lane, M=r; lane>=16 -> N=lane-16, M=r+8.
// ---------------------------------------------------------------------------
template <int K>
__global__ void k_linear_wmma(const float* __restrict__ A, const float* __restrict__ W,
                              const float* __restrict__ bias, const float* __restrict__ resid,
                              float* __restrict__ C, int M, int N, int act) {
  const int wave = (int)blockIdx.x * ((int)blockDim.x >> 5) + ((int)threadIdx.x >> 5);
  const int nT = (N + 15) >> 4, rT = (M + 15) >> 4;
  if (wave >= nT * rT) return;
  const int rt = wave / nT, nt = wave - rt * nT;
  const int row0 = rt << 4, n0 = nt << 4;
  const int lane = (int)threadIdx.x & 31;
  const int half = lane >> 4;
  const int lidx = lane & 15;

  int ar = row0 + lidx; if (ar >= M) ar = M - 1;  // clamp only; never stored
  int bn = n0 + lidx;   if (bn >= N) bn = N - 1;
  const float* ap = A + (size_t)ar * K + half * 8;
  const float* bp = W + (size_t)bn * K + half * 8;

  v8f acc = {};
#pragma unroll
  for (int k0 = 0; k0 < K; k0 += 32) {
    v16h af, bf;
#pragma unroll
    for (int i = 0; i < 8; ++i) {
      af[i] = (_Float16)ap[k0 + i];
      bf[i] = (_Float16)bp[k0 + i];
      if (K - k0 > 16) {  // compile-time
        af[i + 8] = (_Float16)ap[k0 + 16 + i];
        bf[i + 8] = (_Float16)bp[k0 + 16 + i];
      } else {
        af[i + 8] = (_Float16)0.f;
        bf[i + 8] = (_Float16)0.f;
      }
    }
    acc = __builtin_amdgcn_wmma_f32_16x16x32_f16(false, af, false, bf,
                                                 (short)0, acc, false, false);
  }

  const int n = n0 + lidx;
  const int rbase = row0 + half * 8;
  if (n < N) {
    const float bv = bias ? bias[n] : 0.f;
    float* cp = C + (size_t)rbase * N + n;
    const float* rp = resid ? resid + (size_t)rbase * N + n : nullptr;
    if (rbase + 7 < M) {  // full tile fast path: unguarded stores
#pragma unroll
      for (int r = 0; r < 8; ++r) {
        float v = acc[r] + bv;
        if (resid) v += rp[(size_t)r * N];
        if (act) v = lrelu_(v);
        cp[(size_t)r * N] = v;
      }
    } else {
#pragma unroll
      for (int r = 0; r < 8; ++r) {
        if (rbase + r < M) {
          float v = acc[r] + bv;
          if (resid) v += rp[(size_t)r * N];
          if (act) v = lrelu_(v);
          cp[(size_t)r * N] = v;
        }
      }
    }
  }
}

// ---------------------------------------------------------------------------
// Graph-LN: global sum/sumsq reduction + apply (per-channel affine) + lrelu
// ---------------------------------------------------------------------------
__global__ void k_reduce2(const float* __restrict__ X, long n, float* __restrict__ stats) {
  __shared__ float s0[256], s1[256];
  float a = 0.f, b = 0.f;
  long stride = (long)gridDim.x * blockDim.x;
  for (long i = (long)blockIdx.x * blockDim.x + threadIdx.x; i < n; i += stride) {
    float v = X[i]; a += v; b += v * v;
  }
  s0[threadIdx.x] = a; s1[threadIdx.x] = b;
  __syncthreads();
  for (int off = 128; off > 0; off >>= 1) {
    if ((int)threadIdx.x < off) { s0[threadIdx.x] += s0[threadIdx.x + off]; s1[threadIdx.x] += s1[threadIdx.x + off]; }
    __syncthreads();
  }
  if (threadIdx.x == 0) { atomicAdd(&stats[0], s0[0]); atomicAdd(&stats[1], s1[0]); }
}

__global__ void k_graphln(const float* __restrict__ X, float* __restrict__ Y,
                          const float* __restrict__ stats, long n, int C,
                          const float* __restrict__ w, const float* __restrict__ b, int act) {
  long i = (long)blockIdx.x * blockDim.x + threadIdx.x;
  if (i >= n) return;
  float mu  = stats[0] / (float)n;
  float var = stats[1] / (float)n - mu * mu;
  float rs  = rsqrtf(var + 1e-5f);
  int c = (int)(i % C);
  float v = (X[i] - mu) * rs * w[c] + b[c];
  if (act) v = lrelu_(v);
  Y[i] = v;
}

// ---------------------------------------------------------------------------
// Row-LN (per-row over C channels), optional lrelu
// ---------------------------------------------------------------------------
__global__ void k_rowln(const float* __restrict__ X, float* __restrict__ Y,
                        const float* __restrict__ w, const float* __restrict__ b,
                        int M, int C, int act) {
  int r = blockIdx.x * blockDim.x + threadIdx.x;
  if (r >= M) return;
  const float* x = X + (size_t)r * C;
  float mu = 0.f;
  for (int c = 0; c < C; ++c) mu += x[c];
  mu /= (float)C;
  float var = 0.f;
  for (int c = 0; c < C; ++c) { float d = x[c] - mu; var += d * d; }
  var /= (float)C;
  float rs = rsqrtf(var + 1e-5f);
  float* y = Y + (size_t)r * C;
  for (int c = 0; c < C; ++c) {
    float v = (x[c] - mu) * rs * w[c] + b[c];
    if (act) v = lrelu_(v);
    y[c] = v;
  }
}

// ---------------------------------------------------------------------------
// Edge embed: ea2 = lrelu(row_ln(edge_attr @ eW^T + eb))   (8 -> 2 channels)
// ---------------------------------------------------------------------------
__global__ void k_edge_embed(const float* __restrict__ EA, const float* __restrict__ eW,
                             const float* __restrict__ eb, const float* __restrict__ lw,
                             const float* __restrict__ lb, float* __restrict__ out, int E) {
  int e = blockIdx.x * blockDim.x + threadIdx.x;
  if (e >= E) return;
  const float* a = EA + (size_t)e * 8;
  float y0 = eb[0], y1 = eb[1];
#pragma unroll
  for (int k = 0; k < 8; ++k) { y0 += eW[k] * a[k]; y1 += eW[8 + k] * a[k]; }
  float mu = 0.5f * (y0 + y1);
  float d0 = y0 - mu, d1 = y1 - mu;
  float rs = rsqrtf(0.5f * (d0 * d0 + d1 * d1) + 1e-5f);
  out[(size_t)e * 2]     = lrelu_(d0 * rs * lw[0] + lb[0]);
  out[(size_t)e * 2 + 1] = lrelu_(d1 * rs * lw[1] + lb[1]);
}

// self-loop attr = mean of incoming edge attrs (fill_value='mean')
__global__ void k_loop_accum(const int* __restrict__ dst, const float* __restrict__ ea2,
                             float* __restrict__ lcnt, float* __restrict__ lsum, int E) {
  int e = blockIdx.x * blockDim.x + threadIdx.x;
  if (e >= E) return;
  int d = dst[e];
  atomicAdd(&lcnt[d], 1.f);
  atomicAdd(&lsum[(size_t)d * 2],     ea2[(size_t)e * 2]);
  atomicAdd(&lsum[(size_t)d * 2 + 1], ea2[(size_t)e * 2 + 1]);
}
__global__ void k_loop_final(const float* __restrict__ lcnt, const float* __restrict__ lsum,
                             float* __restrict__ loopA, int n) {
  int i = blockIdx.x * blockDim.x + threadIdx.x;
  if (i >= n) return;
  float c = fmaxf(lcnt[i], 1.f);
  loopA[(size_t)i * 2]     = lsum[(size_t)i * 2] / c;
  loopA[(size_t)i * 2 + 1] = lsum[(size_t)i * 2 + 1] / c;
}

// ---------------------------------------------------------------------------
// GATv2 edge passes (edges e<E are real, e>=E are self loops)
// ---------------------------------------------------------------------------
__global__ void k_gat_logit(const float* __restrict__ xl, const float* __restrict__ xr,
                            const float* __restrict__ ea2, const float* __restrict__ loopA,
                            const float* __restrict__ We, const float* __restrict__ att,
                            const int* __restrict__ src, const int* __restrict__ dst,
                            int E, int n, float* __restrict__ logit, unsigned* __restrict__ mmax) {
  int e = blockIdx.x * blockDim.x + threadIdx.x;
  if (e >= E + n) return;
  int s, d; float a0, a1;
  if (e < E) { s = src[e]; d = dst[e]; a0 = ea2[(size_t)e * 2]; a1 = ea2[(size_t)e * 2 + 1]; }
  else       { s = d = e - E; a0 = loopA[(size_t)s * 2]; a1 = loopA[(size_t)s * 2 + 1]; }
  const float* pl = xl + (size_t)s * 64;
  const float* pr = xr + (size_t)d * 64;
  float lg = 0.f;
#pragma unroll 8
  for (int c = 0; c < 64; ++c) {
    float v = pl[c] + pr[c] + We[c * 2] * a0 + We[c * 2 + 1] * a1;
    lg += att[c] * lrelu_(v);
  }
  logit[e] = lg;
  atomicMax(&mmax[d], f2ord(lg));
}

__global__ void k_gat_exp(const float* __restrict__ logit, const unsigned* __restrict__ mmax,
                          const int* __restrict__ dst, int E, int n,
                          float* __restrict__ exv, float* __restrict__ den) {
  int e = blockIdx.x * blockDim.x + threadIdx.x;
  if (e >= E + n) return;
  int d = (e < E) ? dst[e] : e - E;
  float ex = __expf(logit[e] - ord2f(mmax[d]));
  exv[e] = ex;
  atomicAdd(&den[d], ex);
}

__global__ void k_fill_bias(float* __restrict__ out, const float* __restrict__ bias, long n) {
  long i = (long)blockIdx.x * blockDim.x + threadIdx.x;
  if (i < n) out[i] = bias[i & 63];
}

__global__ void k_gat_scatter(const float* __restrict__ xl, const float* __restrict__ exv,
                              const float* __restrict__ den, const int* __restrict__ src,
                              const int* __restrict__ dst, int E, int n, float* __restrict__ out) {
  long t = (long)blockIdx.x * blockDim.x + threadIdx.x;
  int e = (int)(t >> 3);
  if (e >= E + n) return;
  int c0 = ((int)t & 7) * 8;
  int s, d;
  if (e < E) { s = src[e]; d = dst[e]; } else { s = d = e - E; }
  float alpha = exv[e] / den[d];
  const float* pl = xl + (size_t)s * 64 + c0;
  float* po = out + (size_t)d * 64 + c0;
#pragma unroll
  for (int c = 0; c < 8; ++c) atomicAdd(&po[c], pl[c] * alpha);
}

// ---------------------------------------------------------------------------
// MHA core: head_dim=4, SEQ=128; K/V staged in LDS as f16. 1 thread=(node,head)
// ---------------------------------------------------------------------------
__global__ void k_attn(const float* __restrict__ qh, const float* __restrict__ kh,
                       const float* __restrict__ vh, float* __restrict__ out, int M) {
  __shared__ _Float16 sk[TSEQ * 64];
  __shared__ _Float16 sv[TSEQ * 64];
  for (int i = threadIdx.x; i < TSEQ * 64; i += blockDim.x) {
    sk[i] = (_Float16)kh[i];
    sv[i] = (_Float16)vh[i];
  }
  __syncthreads();
  long t = (long)blockIdx.x * blockDim.x + threadIdx.x;
  int node = (int)(t >> 4), head = (int)t & 15;
  if (node >= M) return;
  const int h4 = head * 4;
  const float* q = qh + (size_t)node * 64 + h4;
  float q0 = q[0] * 0.5f, q1 = q[1] * 0.5f, q2 = q[2] * 0.5f, q3 = q[3] * 0.5f; // 1/sqrt(4)
  float m = -3.4e38f;
  for (int s = 0; s < TSEQ; ++s) {
    const _Float16* k = sk + s * 64 + h4;
    float dt = q0 * (float)k[0] + q1 * (float)k[1] + q2 * (float)k[2] + q3 * (float)k[3];
    m = fmaxf(m, dt);
  }
  float sum = 0.f, a0 = 0.f, a1 = 0.f, a2 = 0.f, a3 = 0.f;
  for (int s = 0; s < TSEQ; ++s) {
    const _Float16* k = sk + s * 64 + h4;
    const _Float16* v = sv + s * 64 + h4;
    float dt = q0 * (float)k[0] + q1 * (float)k[1] + q2 * (float)k[2] + q3 * (float)k[3];
    float ex = __expf(dt - m);
    sum += ex;
    a0 += ex * (float)v[0]; a1 += ex * (float)v[1];
    a2 += ex * (float)v[2]; a3 += ex * (float)v[3];
  }
  float* o = out + (size_t)node * 64 + h4;
  float r = 1.f / sum;
  o[0] = a0 * r; o[1] = a1 * r; o[2] = a2 * r; o[3] = a3 * r;
}

// ---------------------------------------------------------------------------
// Mean pooling by graph id
// ---------------------------------------------------------------------------
__global__ void k_pool_scatter(const float* __restrict__ h, const int* __restrict__ batch,
                               float* __restrict__ pooled, float* __restrict__ pcnt, int n) {
  long t = (long)blockIdx.x * blockDim.x + threadIdx.x;
  int i = (int)(t >> 6), c = (int)t & 63;
  if (i >= n) return;
  int g = batch[i];
  atomicAdd(&pooled[(size_t)g * 64 + c], h[t]);
  if (c == 0) atomicAdd(&pcnt[g], 1.f);
}
__global__ void k_pool_div(float* __restrict__ pooled, const float* __restrict__ pcnt, int n) {
  int t = blockIdx.x * blockDim.x + threadIdx.x;
  if (t >= n * 64) return;
  pooled[t] /= fmaxf(pcnt[t >> 6], 1.f);
}

// ===========================================================================
extern "C" void kernel_launch(void* const* d_in, const int* in_sizes, int n_in,
                              void* d_out, int out_size, void* d_ws, size_t ws_size,
                              hipStream_t stream) {
  const float* x    = (const float*)d_in[0];
  const int*   eidx = (const int*)d_in[1];     // [2, NE]: src then dst
  const float* eatt = (const float*)d_in[2];
  const int*   bat  = (const int*)d_in[3];
  const float* text = (const float*)d_in[4];
  // params flattened in JAX pytree (alphabetical, recursive) order after arrays:
  // 5..14 ca1{Wd,Wi,Wo,bd,bi,bo,ln1b,ln1w,ln2b,ln2w}, 15..24 ca2{...},
  // 25..31 conv1{We,Wl,Wr,att,bias,bl,br}, 32..38 conv2{...},
  // 39 dW, 40 db, 41 eW, 42 eb, 43 elnb, 44 elnw,
  // 45 ln1b, 46 ln1w, 47 ln2b, 48 ln2w, 49 lnfb, 50 lnfw,
  // 51 xW, 52 xb, 53 xlnb, 54 xlnw
#define PF(i) ((const float*)d_in[(i)])
  const int* src = eidx;
  const int* dst = eidx + NE;

  float* ws = (float*)d_ws;
  size_t o = 0;
  float* h     = ws + o; o += (size_t)NN * 64;
  float* t0    = ws + o; o += (size_t)NN * 64;
  float* t1    = ws + o; o += (size_t)NN * 64;
  float* t2    = ws + o; o += (size_t)NN * 64;
  float* ea2   = ws + o; o += (size_t)NE * 2;
  float* loopA = ws + o; o += (size_t)NN * 2;
  float* lcnt  = ws + o; o += NN;
  float* lsum  = ws + o; o += (size_t)NN * 2;
  float* logit = ws + o; o += (size_t)(NE + NN);
  float* exv   = ws + o; o += (size_t)(NE + NN);
  float* mmaxf = ws + o; o += NN;
  float* den   = ws + o; o += NN;
  float* kh    = ws + o; o += (size_t)TSEQ * 64;
  float* vh    = ws + o; o += (size_t)TSEQ * 64;
  float* pool  = ws + o; o += (size_t)NG * 64;
  float* pcnt  = ws + o; o += NG;
  float* stats = ws + o; o += 2;
  unsigned* mmax = (unsigned*)mmaxf;

  const int ET = NE + NN;

  auto LIN = [&](const float* A, const float* W, const float* bias, const float* resid,
                 float* C, int M, int K, int N, int act) {
    int waves  = ((M + 15) >> 4) * ((N + 15) >> 4);
    int blocks = (waves + 3) >> 2;
    if (K == 16)
      k_linear_wmma<16><<<blocks, 128, 0, stream>>>(A, W, bias, resid, C, M, N, act);
    else
      k_linear_wmma<64><<<blocks, 128, 0, stream>>>(A, W, bias, resid, C, M, N, act);
  };
  auto GLN = [&](const float* X, float* Y, long n, int C, const float* w, const float* b, int act) {
    k_zero<<<1, 64, 0, stream>>>(stats, 2);
    k_reduce2<<<512, 256, 0, stream>>>(X, n, stats);
    k_graphln<<<(int)((n + 255) / 256), 256, 0, stream>>>(X, Y, stats, n, C, w, b, act);
  };

  // ---- node embed: h = lrelu(graph_ln(x @ xW^T + xb))  [NN,16]
  LIN(x, PF(51), PF(52), nullptr, t0, NN, 64, 16, 0);
  GLN(t0, h, (long)NN * 16, 16, PF(54), PF(53), 1);

  // ---- edge embed: ea2 = lrelu(row_ln(edge_attr @ eW^T + eb))  [NE,2]
  k_edge_embed<<<(NE + 255) / 256, 256, 0, stream>>>(eatt, PF(41), PF(42), PF(44), PF(43), ea2, NE);

  // ---- self-loop attrs (fill='mean')
  k_zero<<<256, 256, 0, stream>>>(lcnt, NN);
  k_zero<<<256, 256, 0, stream>>>(lsum, (long)NN * 2);
  k_loop_accum<<<(NE + 255) / 256, 256, 0, stream>>>(dst, ea2, lcnt, lsum, NE);
  k_loop_final<<<(NN + 255) / 256, 256, 0, stream>>>(lcnt, lsum, loopA, NN);

  // ---- two GATv2 convs + graph-LN + lrelu
  auto CONV = [&](const float* hin, int cin, int pb, const float* lnw, const float* lnb) {
    LIN(hin, PF(pb + 1), PF(pb + 5), nullptr, t0, NN, cin, 64, 0);  // xl = h@Wl^T+bl
    LIN(hin, PF(pb + 2), PF(pb + 6), nullptr, t1, NN, cin, 64, 0);  // xr = h@Wr^T+br
    k_zero<<<256, 256, 0, stream>>>(mmaxf, NN);   // 0u encodes -inf under f2ord
    k_zero<<<256, 256, 0, stream>>>(den, NN);
    k_gat_logit<<<(ET + 255) / 256, 256, 0, stream>>>(t0, t1, ea2, loopA, PF(pb + 0), PF(pb + 3),
                                                      src, dst, NE, NN, logit, mmax);
    k_gat_exp<<<(ET + 255) / 256, 256, 0, stream>>>(logit, mmax, dst, NE, NN, exv, den);
    k_fill_bias<<<(int)(((long)NN * 64 + 255) / 256), 256, 0, stream>>>(t2, PF(pb + 4), (long)NN * 64);
    k_gat_scatter<<<(int)(((long)ET * 8 + 255) / 256), 256, 0, stream>>>(t0, exv, den, src, dst, NE, NN, t2);
    GLN(t2, h, (long)NN * 64, 64, lnw, lnb, 1);
  };
  CONV(h, 16, 25, PF(46), PF(45));  // conv1 + ln1
  CONV(h, 64, 32, PF(48), PF(47));  // conv2 + ln2

  // ---- two cross-attention blocks
  auto CA = [&](int pb) {
    const float* Wi = PF(pb + 1); const float* bi = PF(pb + 4);
    LIN(h,    Wi,             bi,       nullptr, t0, NN,   64, 64, 0);   // q
    LIN(text, Wi + 64 * 64,   bi + 64,  nullptr, kh, TSEQ, 64, 64, 0);   // k
    LIN(text, Wi + 128 * 64,  bi + 128, nullptr, vh, TSEQ, 64, 64, 0);   // v
    k_attn<<<(NN + 7) / 8, 128, 0, stream>>>(t0, kh, vh, t1, NN);
    LIN(t1, PF(pb + 2), PF(pb + 5), h, t2, NN, 64, 64, 0);               // Wo + residual
    k_rowln<<<(NN + 255) / 256, 256, 0, stream>>>(t2, h, PF(pb + 7), PF(pb + 6), NN, 64, 0);
    LIN(h, PF(pb + 0), PF(pb + 3), h, t2, NN, 64, 64, 0);                // FFN + residual
    k_rowln<<<(NN + 255) / 256, 256, 0, stream>>>(t2, h, PF(pb + 9), PF(pb + 8), NN, 64, 0);
  };
  CA(5);   // ca1
  CA(15);  // ca2

  // ---- mean pool by graph, final projection + row-LN + lrelu
  k_zero<<<64, 256, 0, stream>>>(pool, (long)NG * 64);
  k_zero<<<1, 64, 0, stream>>>(pcnt, NG);
  k_pool_scatter<<<(int)(((long)NN * 64 + 255) / 256), 256, 0, stream>>>(h, bat, pool, pcnt, NN);
  k_pool_div<<<(NG * 64 + 255) / 256, 256, 0, stream>>>(pool, pcnt, NG);
  LIN(pool, PF(39), PF(40), nullptr, t2, NG, 64, 256, 0);
  k_rowln<<<1, 64, 0, stream>>>(t2, (float*)d_out, PF(50), PF(49), NG, 256, 1);
#undef PF
}
